// GraphGDP_GCN_79766132621914
// MI455X (gfx1250) — matrix operand order repaired and checked by reading, not compiled, not run on hardware
//
#include <hip/hip_runtime.h>
#include <cstdint>

typedef __attribute__((ext_vector_type(16))) __bf16 bf16x16;
typedef __attribute__((ext_vector_type(8)))  __bf16 bf16x8;
typedef __attribute__((ext_vector_type(8)))  float  floatx8;
typedef __attribute__((ext_vector_type(4)))  float  floatx4;
typedef __attribute__((ext_vector_type(2)))  float  floatx2;
typedef __attribute__((ext_vector_type(4)))  unsigned int u32x4;
typedef __attribute__((ext_vector_type(8)))  int    i32x8;
typedef __attribute__((ext_vector_type(4)))  int    i32x4;
typedef __attribute__((ext_vector_type(4)))  unsigned int uint4v;

#define CCH 128      // channels
#define PE  16
#define CN  112      // C - PE
#define NDIM 422
#define KPAD 448     // NDIM padded to multiple of 32
#define HH  64       // head hidden
#define NG  16       // graphs
#define BNE 1e-5f

// ---------- Tensor Data Mover: 1-D linear copy global -> LDS (bytes multiple of 8) ----------
__device__ static inline void tdm_load_lds(uint32_t lds_off, const void* gptr, uint32_t bytes) {
  uint64_t ga = (uint64_t)(uintptr_t)gptr;
  uint32_t n8 = bytes >> 3;                 // number of 8-byte elements
  u32x4 g0;
  g0[0] = 1u;                               // count=1, user-mode descriptor
  g0[1] = lds_off;                          // lds_addr (bytes)
  g0[2] = (uint32_t)ga;                     // global_addr[31:0]
  g0[3] = (uint32_t)(ga >> 32) | (2u << 30);// global_addr[56:32] | type=2
  i32x8 g1;
  g1[0] = (int)(3u << 16);                  // workgroup_mask=0, data_size=3 (8B)
  g1[1] = (int)(n8 << 16);                  // tensor_dim0[15:0] @ bits 63:48
  g1[2] = (int)(1u << 16);                  // tensor_dim0[31:16]=0, tensor_dim1=1
  g1[3] = (int)(n8 << 16);                  // tile_dim0 @ bits 127:112
  g1[4] = 0;                                // tile_dim1=0, tile_dim2=0 (1-D)
  g1[5] = (int)n8;                          // tensor_dim0_stride[31:0]
  g1[6] = 0;
  g1[7] = 0;
  i32x4 z4 = {0, 0, 0, 0};
#if defined(__clang_major__) && (__clang_major__ >= 23)
  i32x8 z8 = {0, 0, 0, 0, 0, 0, 0, 0};
  __builtin_amdgcn_tensor_load_to_lds(g0, g1, z4, z4, z8, 0);
#else
  __builtin_amdgcn_tensor_load_to_lds(g0, g1, z4, z4, 0);
#endif
}

// ---------------- prep: transposed bf16 weight copies + zero small accumulators ----------------
__global__ void prep_kernel(const float* __restrict__ cw, const float* __restrict__ ew,
                            const float* __restrict__ w1,
                            __bf16* __restrict__ wbT, __bf16* __restrict__ ebT,
                            __bf16* __restrict__ n1bT,
                            float* bnsum, float* bnsum2, float* cnt, float* gsum) {
  int idx = blockIdx.x * 256 + threadIdx.x;
  if (idx < 4 * CCH * CCH) {                 // conv W: [l][k][n] -> [l][n][k]
    int l = idx >> 14, r = idx & 16383;
    int n = r >> 7, k = r & 127;
    wbT[idx] = (__bf16)cw[(l << 14) + k * CCH + n];
  }
  if (idx < CN * KPAD) {                     // emb W: [k][n] -> [n][k], zero pad k>=NDIM
    int n = idx / KPAD, k = idx - n * KPAD;
    ebT[idx] = (__bf16)((k < NDIM) ? ew[k * CN + n] : 0.0f);
  }
  if (idx < HH * CCH) {                      // nh_w1: [k][n] -> [n][k]
    int n = idx >> 7, k = idx & 127;
    n1bT[idx] = (__bf16)w1[k * HH + n];
  }
  if (idx < HH) { bnsum[idx] = 0.0f; bnsum2[idx] = 0.0f; }
  if (idx < NG) cnt[idx] = 0.0f;
  if (idx < NG * CCH) gsum[idx] = 0.0f;
}

// ---------------- LapPE batch statistics (one block per PE dim) ----------------
__global__ void lap_stats_kernel(const float* __restrict__ pe, float* lapm, float* lapi, int N) {
  __shared__ float ss[256], ss2[256];
  int c = blockIdx.x;
  float s = 0.f, s2 = 0.f;
  for (int i = threadIdx.x; i < N; i += 256) {
    float v = pe[(size_t)i * PE + c]; s += v; s2 += v * v;
  }
  ss[threadIdx.x] = s; ss2[threadIdx.x] = s2; __syncthreads();
  for (int off = 128; off; off >>= 1) {
    if (threadIdx.x < off) { ss[threadIdx.x] += ss[threadIdx.x + off]; ss2[threadIdx.x] += ss2[threadIdx.x + off]; }
    __syncthreads();
  }
  if (threadIdx.x == 0) {
    float m = ss[0] / N;
    float var = ss2[0] / N - m * m;
    lapm[c] = m; lapi[c] = rsqrtf(var + BNE);
  }
}

// helper: build bf16 A-fragment from 2x contiguous 8-float runs at p and p+16
__device__ static inline bf16x16 a_frag_f2(const float* p) {   // 8-byte aligned
  bf16x16 a;
#pragma unroll
  for (int q = 0; q < 4; ++q) {
    floatx2 lo = *(const floatx2*)(p + 2 * q);
    floatx2 hi = *(const floatx2*)(p + 16 + 2 * q);
    a[2 * q]     = (__bf16)lo[0];
    a[2 * q + 1] = (__bf16)lo[1];
    a[8 + 2 * q]     = (__bf16)hi[0];
    a[8 + 2 * q + 1] = (__bf16)hi[1];
  }
  return a;
}
__device__ static inline bf16x16 a_frag_f4(const float* p) {   // 16-byte aligned
  floatx4 a0 = *(const floatx4*)(p);
  floatx4 a1 = *(const floatx4*)(p + 4);
  floatx4 a2 = *(const floatx4*)(p + 16);
  floatx4 a3 = *(const floatx4*)(p + 20);
  bf16x16 a;
#pragma unroll
  for (int q = 0; q < 4; ++q) {
    a[q]      = (__bf16)a0[q];
    a[4 + q]  = (__bf16)a1[q];
    a[8 + q]  = (__bf16)a2[q];
    a[12 + q] = (__bf16)a3[q];
  }
  return a;
}

// ---------------- embedding: h = concat(x @ We + be, BN(pe) @ Wl + bl) ----------------
__global__ void embed_gemm_kernel(const float* __restrict__ x, const __bf16* __restrict__ ebT,
                                  const float* __restrict__ eb,
                                  const float* __restrict__ lap_pe, const float* __restrict__ lapm,
                                  const float* __restrict__ lapi, const float* __restrict__ lap_g,
                                  const float* __restrict__ lap_b, const float* __restrict__ lap_w,
                                  const float* __restrict__ lap_bias,
                                  float* __restrict__ h, int N) {
  int m0 = blockIdx.x * 16;
  int wave = threadIdx.x >> 5, lane = threadIdx.x & 31;
  if (wave < 7) {
    int n0 = wave * 16;
    int col = n0 + (lane & 15);
    floatx8 cacc;
    float b = eb[col];
#pragma unroll
    for (int r = 0; r < 8; ++r) cacc[r] = b;
    int row = m0 + (lane & 15);
    int rowc = row < N ? row : (N - 1);            // clamp: rows>=N only feed unstored outputs
    const float* xr = x + (size_t)rowc * NDIM;
    const __bf16* bcol = ebT + (size_t)col * KPAD;
    int ahalf = (lane >> 4) * 8, bhalf = (lane >> 4) * 8;
    // 13 full K-tiles, unconditional vector loads
    for (int kk = 0; kk < 416; kk += 32) {
      bf16x16 afr = a_frag_f2(xr + kk + ahalf);
      bf16x16 bfr = __builtin_shufflevector(*(const bf16x8*)(bcol + kk + bhalf), *(const bf16x8*)(bcol + kk + 16 + bhalf), 0, 1, 2, 3, 4, 5, 6, 7, 8, 9, 10, 11, 12, 13, 14, 15);
      cacc = __builtin_amdgcn_wmma_f32_16x16x32_bf16(false, afr, false, bfr, (short)0, cacc, false, false);
    }
    { // K tail: B rows k>=NDIM are zero-padded, so A only needs in-bounds addresses
      bf16x16 afr, bfr = __builtin_shufflevector(*(const bf16x8*)(bcol + 416 + bhalf), *(const bf16x8*)(bcol + 432 + bhalf), 0, 1, 2, 3, 4, 5, 6, 7, 8, 9, 10, 11, 12, 13, 14, 15);
#pragma unroll
      for (int j = 0; j < 16; ++j) {
        int k = 416 + ahalf + (j < 8 ? j : j + 8);
        int kc = k < NDIM ? k : (NDIM - 1);
        afr[j] = (__bf16)xr[kc];
      }
      cacc = __builtin_amdgcn_wmma_f32_16x16x32_bf16(false, afr, false, bfr, (short)0, cacc, false, false);
    }
#pragma unroll
    for (int r = 0; r < 8; ++r) {
      int orow = m0 + r + (lane >> 4) * 8;
      if (orow < N) h[(size_t)orow * CCH + col] = cacc[r];
    }
  } else {
    // LapPE branch: 16x16 per block, scalar
    int col = lane & 15;
    int rbase = (lane >> 4) * 8;
#pragma unroll
    for (int r = 0; r < 8; ++r) {
      int row = m0 + rbase + r;
      if (row < N) {
        float acc2 = lap_bias[col];
#pragma unroll
        for (int k = 0; k < PE; ++k) {
          float p = (lap_pe[(size_t)row * PE + k] - lapm[k]) * lapi[k] * lap_g[k] + lap_b[k];
          acc2 += p * lap_w[k * PE + col];
        }
        h[(size_t)row * CCH + CN + col] = acc2;
      }
    }
  }
}

// ---------------- degree / dinv ----------------
__global__ void deg_init_kernel(float* dinv, int N) {
  int i = blockIdx.x * 256 + threadIdx.x;
  if (i < N) dinv[i] = 1.0f;   // self loop
}
__global__ void deg_accum_kernel(const int* __restrict__ dst, float* dinv, int E) {
  int e = blockIdx.x * 256 + threadIdx.x;
  if (e < E) atomicAdd(&dinv[dst[e]], 1.0f);
}
__global__ void deg_finish_kernel(float* dinv, int N) {
  int i = blockIdx.x * 256 + threadIdx.x;
  if (i < N) dinv[i] = rsqrtf(dinv[i]);   // deg >= 1 always
}

// ---------------- conv GEMM: t = h @ W_l ; W^T staged in LDS via TDM ----------------
__global__ void conv_gemm_kernel(const float* __restrict__ h, const __bf16* __restrict__ wbT,
                                 float* __restrict__ out, int N) {
  __shared__ float  sA[16 * 132];            // padded (132*4 % 64 banks spreads rows)
  __shared__ __bf16 sWT[CCH * CCH];          // [n][k] transposed
  int m0 = blockIdx.x * 16;
  int tid = threadIdx.x;

  if (tid == 0) {
    tdm_load_lds((uint32_t)(uintptr_t)&sWT[0], wbT, CCH * CCH * (uint32_t)sizeof(__bf16));
  }
  // stage A tile (16 rows x 128 cols) as float4, rows>=N zero-filled
  const floatx4 vz = {0.f, 0.f, 0.f, 0.f};
  for (int idx = tid; idx < 16 * 32; idx += 256) {
    int r = idx >> 5, c4 = (idx & 31) * 4;
    int row = m0 + r;
    floatx4 v = (row < N) ? *(const floatx4*)(h + (size_t)row * CCH + c4) : vz;
    *(floatx4*)(sA + r * 132 + c4) = v;
  }
  __builtin_amdgcn_s_wait_tensorcnt(0);      // wave 0 waits for TDM; others pass
  __syncthreads();

  int wave = tid >> 5, lane = tid & 31;
  int n0 = wave * 16;
  int col = n0 + (lane & 15);
  floatx8 cacc;
#pragma unroll
  for (int r = 0; r < 8; ++r) cacc[r] = 0.0f;
  int arow = lane & 15;
  int ahalf = (lane >> 4) * 8, bhalf = (lane >> 4) * 8;
  const float* ap = sA + arow * 132;
  const __bf16* bp = sWT + (size_t)col * CCH;
#pragma unroll
  for (int kk = 0; kk < CCH; kk += 32) {
    bf16x16 afr = a_frag_f4(ap + kk + ahalf);
    bf16x16 bfr = __builtin_shufflevector(*(const bf16x8*)(bp + kk + bhalf), *(const bf16x8*)(bp + kk + 16 + bhalf), 0, 1, 2, 3, 4, 5, 6, 7, 8, 9, 10, 11, 12, 13, 14, 15);
    cacc = __builtin_amdgcn_wmma_f32_16x16x32_bf16(false, afr, false, bfr, (short)0, cacc, false, false);
  }
#pragma unroll
  for (int r = 0; r < 8; ++r) {
    int row = m0 + r + (lane >> 4) * 8;
    if (row < N) out[(size_t)row * CCH + col] = cacc[r];
  }
}

// ---------------- bias init of scatter accumulator ----------------
__global__ void acc_init_kernel(float* acc, const float* __restrict__ cb, int NC) {
  int i = blockIdx.x * 256 + threadIdx.x;
  if (i < NC) acc[i] = cb[i & 127];
}

// ---------------- edge scatter: acc[dst] += t[src] * dinv[src]*dinv[dst]  (wave per edge) --
__global__ void scatter_kernel(const float* __restrict__ t, const int* __restrict__ src,
                               const int* __restrict__ dst, const float* __restrict__ dinv,
                               float* acc, int E, int N) {
  int wv = (blockIdx.x * blockDim.x + threadIdx.x) >> 5;
  int lane = threadIdx.x & 31;
  int total = E + N;
  if (wv >= total) return;
  int s, d; float nrm;
  if (wv < E) {
    s = src[wv]; d = dst[wv];
    nrm = dinv[s] * dinv[d];
  } else {
    s = d = wv - E;
    float di = dinv[s];
    nrm = di * di;
  }
  __builtin_prefetch(t + (size_t)s * CCH + lane * 4, 0, 0);
  floatx4 v = *(const floatx4*)(t + (size_t)s * CCH + lane * 4);
  float* ap = acc + (size_t)d * CCH + lane * 4;
  atomicAdd(ap + 0, v[0] * nrm);
  atomicAdd(ap + 1, v[1] * nrm);
  atomicAdd(ap + 2, v[2] * nrm);
  atomicAdd(ap + 3, v[3] * nrm);
}

// ---------------- ReLU copy ----------------
__global__ void relu_kernel(const float* __restrict__ acc, float* __restrict__ h, int NC) {
  int i = blockIdx.x * 256 + threadIdx.x;
  if (i < NC) h[i] = fmaxf(acc[i], 0.0f);
}

// ---------------- node head GEMM: z = h @ nh_w1 + b1 (N x 64) ----------------
__global__ void head_gemm1_kernel(const float* __restrict__ h, const __bf16* __restrict__ w1bT,
                                  const float* __restrict__ b1, float* __restrict__ z, int N) {
  int wave = threadIdx.x >> 5, lane = threadIdx.x & 31;
  int m0 = blockIdx.x * 32 + (wave >> 2) * 16;
  int n0 = (wave & 3) * 16;
  int col = n0 + (lane & 15);
  floatx8 cacc;
  float b = b1[col];
#pragma unroll
  for (int r = 0; r < 8; ++r) cacc[r] = b;
  int row = m0 + (lane & 15);
  int rowc = row < N ? row : (N - 1);
  const float* hr = h + (size_t)rowc * CCH;
  const __bf16* bcol = w1bT + (size_t)col * CCH;
  int ahalf = (lane >> 4) * 8, bhalf = (lane >> 4) * 8;
#pragma unroll
  for (int kk = 0; kk < CCH; kk += 32) {
    bf16x16 afr = a_frag_f4(hr + kk + ahalf);
    bf16x16 bfr = __builtin_shufflevector(*(const bf16x8*)(bcol + kk + bhalf), *(const bf16x8*)(bcol + kk + 16 + bhalf), 0, 1, 2, 3, 4, 5, 6, 7, 8, 9, 10, 11, 12, 13, 14, 15);
    cacc = __builtin_amdgcn_wmma_f32_16x16x32_bf16(false, afr, false, bfr, (short)0, cacc, false, false);
  }
#pragma unroll
  for (int r = 0; r < 8; ++r) {
    int orow = m0 + r + (lane >> 4) * 8;
    if (orow < N) z[(size_t)orow * HH + col] = cacc[r];
  }
}

// ---------------- BN stats over z (N x 64) ----------------
__global__ void bn_stats_kernel(const float* __restrict__ z, float* bnsum, float* bnsum2, int N) {
  __shared__ float ss[256], ss2[256];
  int tid = threadIdx.x;
  int c = tid & 63;
  int rl = tid >> 6;       // 0..3
  float s = 0.f, s2 = 0.f;
  for (int r = blockIdx.x * 4 + rl; r < N; r += gridDim.x * 4) {
    float v = z[(size_t)r * HH + c]; s += v; s2 += v * v;
  }
  ss[tid] = s; ss2[tid] = s2; __syncthreads();
  if (rl == 0) {
    s  = ss[c]  + ss[64 + c]  + ss[128 + c]  + ss[192 + c];
    s2 = ss2[c] + ss2[64 + c] + ss2[128 + c] + ss2[192 + c];
    atomicAdd(&bnsum[c], s);
    atomicAdd(&bnsum2[c], s2);
  }
}
__global__ void bn_final_kernel(float* bnsum, float* bnsum2, int N) {
  int c = threadIdx.x;
  if (c < HH) {
    float m = bnsum[c] / N;
    float var = bnsum2[c] / N - m * m;
    bnsum[c] = m;                       // reuse as mean
    bnsum2[c] = rsqrtf(var + BNE);      // reuse as inv-std
  }
}

// ---------------- node head part 2: GELU + Linear(64->2), wave per node ----------------
__global__ void node_head2_kernel(const float* __restrict__ z, const float* __restrict__ m,
                                  const float* __restrict__ istd, const float* __restrict__ g,
                                  const float* __restrict__ be, const float* __restrict__ w2,
                                  const float* __restrict__ b2, float* __restrict__ out, int N) {
  int wv = (blockIdx.x * blockDim.x + threadIdx.x) >> 5;
  int lane = threadIdx.x & 31;
  if (wv >= N) return;
  int k = lane * 2;
  floatx2 v = *(const floatx2*)(z + (size_t)wv * HH + k);
  float a0 = (v[0] - m[k]) * istd[k] * g[k] + be[k];
  float a1 = (v[1] - m[k + 1]) * istd[k + 1] * g[k + 1] + be[k + 1];
  a0 = 0.5f * a0 * (1.0f + erff(a0 * 0.70710678118f));
  a1 = 0.5f * a1 * (1.0f + erff(a1 * 0.70710678118f));
  float p0 = a0 * w2[k * 2]     + a1 * w2[(k + 1) * 2];
  float p1 = a0 * w2[k * 2 + 1] + a1 * w2[(k + 1) * 2 + 1];
  for (int off = 16; off; off >>= 1) {
    p0 += __shfl_xor(p0, off, 32);
    p1 += __shfl_xor(p1, off, 32);
  }
  if (lane == 0) {
    out[(size_t)wv * 2]     = p0 + b2[0];
    out[(size_t)wv * 2 + 1] = p1 + b2[1];
  }
}

// ---------------- per-graph pooling (wave per node) ----------------
__global__ void pool_kernel(const float* __restrict__ h, const int* __restrict__ batch,
                            float* cnt, float* gsum, int N) {
  int wv = (blockIdx.x * blockDim.x + threadIdx.x) >> 5;
  int lane = threadIdx.x & 31;
  if (wv >= N) return;
  int b = batch[wv];
  floatx4 v = *(const floatx4*)(h + (size_t)wv * CCH + lane * 4);
  float* gp = gsum + (size_t)b * CCH + lane * 4;
  atomicAdd(gp + 0, v[0]);
  atomicAdd(gp + 1, v[1]);
  atomicAdd(gp + 2, v[2]);
  atomicAdd(gp + 3, v[3]);
  if (lane == 0) atomicAdd(&cnt[b], 1.0f);
}

// ---------------- graph head (single block) ----------------
__global__ void graph_head_kernel(const float* __restrict__ cnt, const float* __restrict__ gsum,
                                  const float* __restrict__ w1, const float* __restrict__ b1,
                                  const float* __restrict__ g, const float* __restrict__ be,
                                  const float* __restrict__ w2, const float* __restrict__ b2,
                                  float* __restrict__ out) {
  __shared__ float z[NG * HH];
  __shared__ float mm[HH], ii[HH];
  int tid = threadIdx.x;
  for (int idx = tid; idx < NG * HH; idx += 256) {
    int gi = idx / HH, c = idx - gi * HH;
    float cg = fmaxf(cnt[gi], 1.0f);
    float acc = b1[c];
    for (int k = 0; k < CCH; ++k) acc += (gsum[gi * CCH + k] / cg) * w1[k * HH + c];
    z[idx] = acc;
  }
  __syncthreads();
  if (tid < HH) {
    float s = 0.f, s2 = 0.f;
    for (int gi = 0; gi < NG; ++gi) { float v = z[gi * HH + tid]; s += v; s2 += v * v; }
    float m = s / NG;
    float var = s2 / NG - m * m;
    mm[tid] = m; ii[tid] = rsqrtf(var + BNE);
  }
  __syncthreads();
  if (tid < NG * 2) {
    int gi = tid >> 1, j = tid & 1;
    float acc = b2[j];
    for (int c = 0; c < HH; ++c) {
      float a = (z[gi * HH + c] - mm[c]) * ii[c] * g[c] + be[c];
      a = 0.5f * a * (1.0f + erff(a * 0.70710678118f));
      acc += a * w2[c * 2 + j];
    }
    out[gi * 2 + j] = acc;
  }
}

extern "C" void kernel_launch(void* const* d_in, const int* in_sizes, int n_in,
                              void* d_out, int out_size, void* d_ws, size_t ws_size,
                              hipStream_t stream) {
  const float* x          = (const float*)d_in[0];
  const int*   ei         = (const int*)d_in[1];
  const int*   batch      = (const int*)d_in[2];
  const float* lap_pe     = (const float*)d_in[3];
  const float* node_emb_w = (const float*)d_in[4];
  const float* node_emb_b = (const float*)d_in[5];
  const float* lap_g      = (const float*)d_in[6];
  const float* lap_b      = (const float*)d_in[7];
  const float* lap_w      = (const float*)d_in[8];
  const float* lap_bias   = (const float*)d_in[9];
  const float* conv_w     = (const float*)d_in[10];
  const float* conv_b     = (const float*)d_in[11];
  const float* nh_w1      = (const float*)d_in[12];
  const float* nh_b1      = (const float*)d_in[13];
  const float* nh_g       = (const float*)d_in[14];
  const float* nh_be      = (const float*)d_in[15];
  const float* nh_w2      = (const float*)d_in[16];
  const float* nh_b2      = (const float*)d_in[17];
  const float* gh_w1      = (const float*)d_in[18];
  const float* gh_b1      = (const float*)d_in[19];
  const float* gh_g       = (const float*)d_in[20];
  const float* gh_be      = (const float*)d_in[21];
  const float* gh_w2      = (const float*)d_in[22];
  const float* gh_b2      = (const float*)d_in[23];

  const int N = in_sizes[2];          // 100000
  const int E = in_sizes[1] / 2;      // 1600000
  const int NC = N * CCH;
  const int* esrc = ei;
  const int* edst = ei + E;

  // workspace layout (floats, then bf16 tail)
  float* wsf   = (float*)d_ws;
  float* h     = wsf;                    // N*C
  float* t     = h + (size_t)NC;         // N*C
  float* acc   = t + (size_t)NC;         // N*C
  float* dinv  = acc + (size_t)NC;       // N
  float* lapm  = dinv + N;               // 16
  float* lapi  = lapm + PE;              // 16
  float* bnsum = lapi + PE;              // 64
  float* bnsum2= bnsum + HH;             // 64
  float* cnt   = bnsum2 + HH;            // 16
  float* gsum  = cnt + NG;               // 16*128
  __bf16* wbT  = (__bf16*)(gsum + NG * CCH);   // 4*128*128 (transposed per layer)
  __bf16* ebT  = wbT + 4 * CCH * CCH;          // 112*448 (transposed, K-padded)
  __bf16* n1bT = ebT + CN * KPAD;              // 64*128 (transposed)

  float* node_out   = (float*)d_out;           // [N,2]
  float* global_out = node_out + (size_t)N * 2;// [16,2]

  const int mtiles = (N + 15) / 16;            // 6250

  prep_kernel<<<(4 * CCH * CCH + 255) / 256, 256, 0, stream>>>(
      conv_w, node_emb_w, nh_w1, wbT, ebT, n1bT, bnsum, bnsum2, cnt, gsum);
  lap_stats_kernel<<<PE, 256, 0, stream>>>(lap_pe, lapm, lapi, N);
  embed_gemm_kernel<<<mtiles, 256, 0, stream>>>(
      x, ebT, node_emb_b, lap_pe, lapm, lapi, lap_g, lap_b, lap_w, lap_bias, h, N);

  deg_init_kernel<<<(N + 255) / 256, 256, 0, stream>>>(dinv, N);
  deg_accum_kernel<<<(E + 255) / 256, 256, 0, stream>>>(edst, dinv, E);
  deg_finish_kernel<<<(N + 255) / 256, 256, 0, stream>>>(dinv, N);

  const int scatter_blocks = (E + N + 7) / 8;  // wave per edge, 8 waves/block
  for (int l = 0; l < 4; ++l) {
    acc_init_kernel<<<(NC + 255) / 256, 256, 0, stream>>>(acc, conv_b + l * CCH, NC);
    conv_gemm_kernel<<<mtiles, 256, 0, stream>>>(h, wbT + (size_t)l * CCH * CCH, t, N);
    scatter_kernel<<<scatter_blocks, 256, 0, stream>>>(t, esrc, edst, dinv, acc, E, N);
    relu_kernel<<<(NC + 255) / 256, 256, 0, stream>>>(acc, h, NC);
  }

  head_gemm1_kernel<<<(N + 31) / 32, 256, 0, stream>>>(h, n1bT, nh_b1, t, N);
  bn_stats_kernel<<<400, 256, 0, stream>>>(t, bnsum, bnsum2, N);
  bn_final_kernel<<<1, 64, 0, stream>>>(bnsum, bnsum2, N);
  node_head2_kernel<<<(N + 7) / 8, 256, 0, stream>>>(
      t, bnsum, bnsum2, nh_g, nh_be, nh_w2, nh_b2, node_out, N);

  pool_kernel<<<(N + 7) / 8, 256, 0, stream>>>(h, batch, cnt, gsum, N);
  graph_head_kernel<<<1, 256, 0, stream>>>(
      cnt, gsum, gh_w1, gh_b1, gh_g, gh_be, gh_w2, gh_b2, global_out);
}
